// DCRNNModel_DoubleEncoder_30159260352640
// MI455X (gfx1250) — matrix operand order repaired and checked by reading, not compiled, not run on hardware
//
#include <hip/hip_runtime.h>
#include <math.h>

// ---------------------------------------------------------------------------
// DCRNN double-encoder for MI455X (gfx1250, wave32, WMMA).
// One block per (encoder, batch) runs the whole T=128 scan with all state in
// LDS. All matmuls use v_wmma_f32_16x16x32_bf16 (f32 accum). K-loops are kept
// rolled (#pragma unroll 1) so the compiler cannot hoist weight B-fragments
// out of the time loop and spill to scratch.
// ---------------------------------------------------------------------------

typedef __attribute__((ext_vector_type(16))) __bf16 v16bf;
typedef __attribute__((ext_vector_type(8)))  __bf16 v8bf;
typedef __attribute__((ext_vector_type(8)))  float  v8f;

// Problem constants
#define TT   128
#define BB   64
#define NN   64
#define DIN  100
#define HH   64
#define CC   4
#define KP0  192   // padded K for layer0 (164 -> 192)
#define KP1  128   // K for layer1
#define YLD  384   // Y buffer leading dim (3*128 max)

// ws layout: [ 2MB last(e,b,n,h) f32 | bf16 weights ]
#define LAST_BYTES (2u*BB*NN*HH*4u)
// per-encoder bf16 weight elements
#define WG0_ELEMS (384*KP0)
#define WC0_ELEMS (192*KP0)
#define WG1_ELEMS (384*KP1)
#define WC1_ELEMS (192*KP1)
#define ENC_W_ELEMS (WG0_ELEMS + WC0_ELEMS + WG1_ELEMS + WC1_ELEMS)  // 184320

#define SMEM_BYTES (64*YLD*4 + 3*4096*4 + (64*KP0 + 128*64 + 64*64)*2) // 196608

// ---------------------------------------------------------------------------
// Fragment loaders (16-bit, 16x16x32); A/B share a contiguous k order, not ISA 7.12.2's table order.
// A (16x32): M=lane&15, elems i -> K = kk + i + 16*(lane>=16) (contiguous).
// B (32x16): N=lane&15, elems i -> K = kk + i + 16*(lane>=16) (contiguous).
// C/D f32:   vgpr r: M = m0 + r + 8*(lane>=16), N = n0 + (lane&15).
// ---------------------------------------------------------------------------
__device__ inline v16bf load_a(const __bf16* base, int ld, int m0, int kk, int lane) {
  int m  = m0 + (lane & 15);
  int k0 = kk + ((lane >> 4) << 4);
  const __bf16* p = base + m * ld + k0;
  union { v16bf v; v8bf h[2]; } u;
  u.h[0] = *(const v8bf*)(p);
  u.h[1] = *(const v8bf*)(p + 8);
  return u.v;
}

__device__ inline v16bf load_b(const __bf16* base, int ld, int n0, int kk, int lane) {
  int n  = n0 + (lane & 15);
  int k0 = kk + ((lane >> 4) << 4);
  return *(const v16bf*)(base + n * ld + k0);
}

#define VZERO8 {0.f,0.f,0.f,0.f,0.f,0.f,0.f,0.f}

// Main GEMM chunk: 1 M-tile x 6 N-tiles, A from LDS Xg, B from global weights
// (pre-transposed [ncol][k] bf16), C written f32 to LDS Y.
// NOTE: k-loop intentionally rolled to bound register pressure.
template<int KP>
__device__ inline void gemm6(const __bf16* __restrict__ Xg,
                             const __bf16* __restrict__ Wt,
                             float* __restrict__ Y,
                             int mt, int nt0, int lane) {
  v8f acc[6] = {VZERO8, VZERO8, VZERO8, VZERO8, VZERO8, VZERO8};
  #pragma unroll 1
  for (int kk = 0; kk < KP; kk += 32) {
    v16bf a = load_a(Xg, KP, mt * 16, kk, lane);
    #pragma unroll
    for (int i = 0; i < 6; ++i) {
      v16bf bf = load_b(Wt, KP, (nt0 + i) * 16, kk, lane);
      acc[i] = __builtin_amdgcn_wmma_f32_16x16x32_bf16(
          false, a, false, bf, (short)0, acc[i], false, false);
    }
  }
  const int nn = lane & 15, hi = lane >> 4;
  #pragma unroll
  for (int i = 0; i < 6; ++i) {
    int n = (nt0 + i) * 16 + nn;
    #pragma unroll
    for (int r = 0; r < 8; ++r)
      Y[(mt * 16 + r + 8 * hi) * YLD + n] = acc[i][r];
  }
}

// ---------------------------------------------------------------------------
// One graph-conv + activation stage.
//   g = (Y0 - Y2) + S*(Y1 + 2*S*Y2) + bias,  Yk = Xg @ Wk
// GATE:  s = sigmoid(g); n<64 -> write bf16(s*h) into Xg[:, xoff+n]; else u.
// !GATE: c = tanh(g); h = u*h + (1-u)*c.
// ---------------------------------------------------------------------------
template<int KP, int NOUT, bool GATE>
__device__ void gconv(const __bf16* __restrict__ Wt, const float* __restrict__ bias,
                      float* __restrict__ Y, float* __restrict__ h,
                      float* __restrict__ uL, __bf16* __restrict__ Xg,
                      __bf16* __restrict__ Ybt, const __bf16* __restrict__ Sbf,
                      int xoff, int tid) {
  const int lane = tid & 31;
  const int w    = tid >> 5;
  const int mt   = w & 3;       // 4 M-tiles (64 rows)
  const int g    = w >> 2;      // 2 wave-groups over N
  constexpr int PER = (3 * NOUT / 16) / 2;  // N-tiles per wave-group: 12 or 6

  // ---- fused main GEMM: Y[:, 0:3*NOUT] = Xg @ [W0|W1|W2]
  #pragma unroll 1
  for (int c = 0; c < PER / 6; ++c)
    gemm6<KP>(Xg, Wt, Y, mt, g * PER + c * 6, lane);
  __syncthreads();

  // ---- stage bf16(Y2) transposed [n][k] for use as S-GEMM B operand
  #pragma unroll 1
  for (int idx = tid; idx < NOUT * 64; idx += 256) {
    int n = idx >> 6, k = idx & 63;
    Ybt[n * 64 + k] = (__bf16)Y[k * YLD + 2 * NOUT + n];
  }
  __syncthreads();

  constexpr int NTS = NOUT / 32;  // S-GEMM N-tiles per wave: 4 or 2
  v8f acc[NTS];

  // ---- P = S * Y2 ; Z = Y1 + 2P (kept in regs)
  #pragma unroll
  for (int i = 0; i < NTS; ++i) {
    v8f t = VZERO8;
    #pragma unroll 1
    for (int kk = 0; kk < 64; kk += 32) {
      v16bf a  = load_a(Sbf, 64, mt * 16, kk, lane);
      v16bf bf = load_b(Ybt, 64, (g * NTS + i) * 16, kk, lane);
      t = __builtin_amdgcn_wmma_f32_16x16x32_bf16(false, a, false, bf, (short)0, t, false, false);
    }
    acc[i] = t;
  }
  {
    const int nn = lane & 15, hi = lane >> 4;
    #pragma unroll
    for (int i = 0; i < NTS; ++i) {
      int n = (g * NTS + i) * 16 + nn;
      #pragma unroll
      for (int r = 0; r < 8; ++r) {
        int m = mt * 16 + r + 8 * hi;
        acc[i][r] = Y[m * YLD + NOUT + n] + 2.f * acc[i][r];
      }
    }
  }
  __syncthreads();  // all waves done reading Ybt(Y2)
  {
    const int nn = lane & 15, hi = lane >> 4;
    #pragma unroll
    for (int i = 0; i < NTS; ++i) {
      int n = (g * NTS + i) * 16 + nn;
      #pragma unroll
      for (int r = 0; r < 8; ++r)
        Ybt[n * 64 + (mt * 16 + r + 8 * hi)] = (__bf16)acc[i][r];
    }
  }
  __syncthreads();

  // ---- Q = S * Z ; finalize g = Q + Y0 - Y2 + bias ; activation + update
  #pragma unroll 1
  for (int i = 0; i < NTS; ++i) {
    v8f q = VZERO8;
    #pragma unroll 1
    for (int kk = 0; kk < 64; kk += 32) {
      v16bf a  = load_a(Sbf, 64, mt * 16, kk, lane);
      v16bf bf = load_b(Ybt, 64, (g * NTS + i) * 16, kk, lane);
      q = __builtin_amdgcn_wmma_f32_16x16x32_bf16(false, a, false, bf, (short)0, q, false, false);
    }
    const int nn = lane & 15, hi = lane >> 4;
    const int n  = (g * NTS + i) * 16 + nn;
    const float bv = bias[n];
    #pragma unroll
    for (int r = 0; r < 8; ++r) {
      int m = mt * 16 + r + 8 * hi;
      float gg = q[r] + Y[m * YLD + n] - Y[m * YLD + 2 * NOUT + n] + bv;
      if (GATE) {
        float s = 1.f / (1.f + __expf(-gg));
        if (n < 64) Xg[m * KP + xoff + n] = (__bf16)(s * h[m * 64 + n]);  // r*h
        else        uL[m * 64 + (n - 64)] = s;                            // u
      } else {
        float c  = tanhf(gg);
        float uu = uL[m * 64 + n];
        h[m * 64 + n] = uu * h[m * 64 + n] + (1.f - uu) * c;
      }
    }
  }
  __syncthreads();
}

// ---------------------------------------------------------------------------
struct EncParams {
  const float* sup;
  const float* bg0; const float* bc0;
  const float* bg1; const float* bc1;
};

__global__ __launch_bounds__(256, 1)
void dcrnn_encoder_kernel(const float* __restrict__ xin, const int* __restrict__ slen,
                          EncParams p0, EncParams p1,
                          const __bf16* __restrict__ wbase, float* __restrict__ last) {
  extern __shared__ char smem[];
  float* Y   = (float*)smem;            // 64 x 384 f32
  float* h0  = Y + 64 * YLD;            // 64 x 64
  float* h1  = h0 + 4096;
  float* uL  = h1 + 4096;
  __bf16* Xg  = (__bf16*)(uL + 4096);   // 64 x 192 bf16
  __bf16* Ybt = Xg + 64 * KP0;          // 128 x 64 bf16 (transposed staging)
  __bf16* Sbf = Ybt + 128 * 64;         // 64 x 64 bf16

  const int tid = threadIdx.x;
  const int e   = blockIdx.x >> 6;
  const int b   = blockIdx.x & 63;
  const EncParams P = e ? p1 : p0;
  const __bf16* wg0 = wbase + (size_t)e * ENC_W_ELEMS;
  const __bf16* wc0 = wg0 + WG0_ELEMS;
  const __bf16* wg1 = wc0 + WC0_ELEMS;
  const __bf16* wc1 = wg1 + WG1_ELEMS;

  for (int i = tid; i < 4096; i += 256) {
    h0[i] = 0.f; h1[i] = 0.f; Sbf[i] = (__bf16)P.sup[i];
  }
  for (int i = tid; i < 64 * (KP0 - 164); i += 256) {   // zero K-padding cols
    int n = i / (KP0 - 164), d = i % (KP0 - 164);
    Xg[n * KP0 + 164 + d] = (__bf16)0.f;
  }
  int tl = slen[b] - 1;
  if (tl < 0) tl = 0; if (tl > TT - 1) tl = TT - 1;
  __syncthreads();

  #pragma unroll 1
  for (int t = 0; t < TT; ++t) {
    // ---- layer 0: Xg = [bf16(x_t) | bf16(h0) | 0pad]
    const float* xt = xin + ((size_t)(b * TT + t)) * NN * DIN;
    #pragma unroll 1
    for (int i = tid; i < NN * DIN; i += 256) {
      int n = i / DIN, d = i % DIN;
      Xg[n * KP0 + d] = (__bf16)xt[i];
    }
    #pragma unroll 1
    for (int i = tid; i < 4096; i += 256) {
      int n = i >> 6, j = i & 63;
      Xg[n * KP0 + DIN + j] = (__bf16)h0[i];
    }
    __syncthreads();
    gconv<KP0, 128, true >(wg0, P.bg0, Y, h0, uL, Xg, Ybt, Sbf, DIN, tid);
    gconv<KP0,  64, false>(wc0, P.bc0, Y, h0, uL, Xg, Ybt, Sbf, DIN, tid);

    // ---- layer 1: Xg = [bf16(h0) | bf16(h1)]
    #pragma unroll 1
    for (int i = tid; i < NN * KP1; i += 256) {
      int n = i >> 7, j = i & 127;
      Xg[n * KP1 + j] = (__bf16)((j < 64) ? h0[n * 64 + j] : h1[n * 64 + (j - 64)]);
    }
    __syncthreads();
    gconv<KP1, 128, true >(wg1, P.bg1, Y, h1, uL, Xg, Ybt, Sbf, 64, tid);
    gconv<KP1,  64, false>(wc1, P.bc1, Y, h1, uL, Xg, Ybt, Sbf, 64, tid);

    if (t == tl) {
      float* dst = last + ((size_t)(e * BB + b)) * NN * HH;
      #pragma unroll 1
      for (int i = tid; i < 4096; i += 256) dst[i] = h1[i];
    }
    __syncthreads();
  }
}

// ---------------------------------------------------------------------------
// Weight pre-pass: de-interleave Chebyshev order k, transpose to [ncol][kp],
// cast to bf16, zero-pad kp>=Dg.  dst[(k*Nout+n)*KP + kp]
// ---------------------------------------------------------------------------
__global__ void conv_w(const float* __restrict__ W, __bf16* __restrict__ dst,
                       int Dg, int Nout, int KP) {
  int total = 3 * Nout * KP;
  for (int idx = blockIdx.x * blockDim.x + threadIdx.x; idx < total;
       idx += gridDim.x * blockDim.x) {
    int kp = idx % KP;
    int rest = idx / KP;
    int n = rest % Nout;
    int k = rest / Nout;
    float v = (kp < Dg) ? W[(kp * 3 + k) * Nout + n] : 0.f;
    dst[idx] = (__bf16)v;
  }
}

// ---------------------------------------------------------------------------
// Final FC + max over nodes: out[b][c] = max_n( relu(last)·fc_w + fc_b )
// ---------------------------------------------------------------------------
__global__ void fc_kernel(const float* __restrict__ last, const float* __restrict__ fcw,
                          const float* __restrict__ fcb, float* __restrict__ out) {
  int b = blockIdx.x, n = threadIdx.x;
  float a[CC] = {fcb[0], fcb[1], fcb[2], fcb[3]};
  const float* l1 = last + ((size_t)(b) * NN + n) * HH;           // encoder 1
  const float* l2 = last + ((size_t)(BB + b) * NN + n) * HH;      // encoder 2
  for (int j = 0; j < HH; ++j) {
    float v1 = fmaxf(l1[j], 0.f);
    float v2 = fmaxf(l2[j], 0.f);
    #pragma unroll
    for (int c = 0; c < CC; ++c)
      a[c] += v1 * fcw[j * CC + c] + v2 * fcw[(HH + j) * CC + c];
  }
  __shared__ float red[NN * CC];
  #pragma unroll
  for (int c = 0; c < CC; ++c) red[n * CC + c] = a[c];
  __syncthreads();
  if (n < CC) {
    float m = red[n];
    for (int i = 1; i < NN; ++i) m = fmaxf(m, red[i * CC + n]);
    out[b * CC + n] = m;
  }
}

// ---------------------------------------------------------------------------
extern "C" void kernel_launch(void* const* d_in, const int* in_sizes, int n_in,
                              void* d_out, int out_size, void* d_ws, size_t ws_size,
                              hipStream_t stream) {
  (void)in_sizes; (void)n_in; (void)out_size; (void)ws_size;
  const float* xin  = (const float*)d_in[0];
  const int*   slen = (const int*)d_in[1];

  float*  last  = (float*)d_ws;
  __bf16* wbase = (__bf16*)((char*)d_ws + LAST_BYTES);

  // Weight conversion: e1 {Wg0,Wc0,Wg1,Wc1} then e2.
  struct Job { int src; int Dg, Nout, KP; size_t off; };
  const Job jobs[8] = {
    { 4, 164, 128, KP0, 0 },
    { 6, 164,  64, KP0, WG0_ELEMS },
    { 8, 128, 128, KP1, WG0_ELEMS + WC0_ELEMS },
    {10, 128,  64, KP1, WG0_ELEMS + WC0_ELEMS + WG1_ELEMS },
    {12, 164, 128, KP0, (size_t)ENC_W_ELEMS },
    {14, 164,  64, KP0, (size_t)ENC_W_ELEMS + WG0_ELEMS },
    {16, 128, 128, KP1, (size_t)ENC_W_ELEMS + WG0_ELEMS + WC0_ELEMS },
    {18, 128,  64, KP1, (size_t)ENC_W_ELEMS + WG0_ELEMS + WC0_ELEMS + WG1_ELEMS },
  };
  for (int j = 0; j < 8; ++j) {
    int total = 3 * jobs[j].Nout * jobs[j].KP;
    conv_w<<<(total + 255) / 256, 256, 0, stream>>>(
        (const float*)d_in[jobs[j].src], wbase + jobs[j].off,
        jobs[j].Dg, jobs[j].Nout, jobs[j].KP);
  }

  EncParams p0{(const float*)d_in[2], (const float*)d_in[5], (const float*)d_in[7],
               (const float*)d_in[9], (const float*)d_in[11]};
  EncParams p1{(const float*)d_in[3], (const float*)d_in[13], (const float*)d_in[15],
               (const float*)d_in[17], (const float*)d_in[19]};

  hipFuncSetAttribute(reinterpret_cast<const void*>(dcrnn_encoder_kernel),
                      hipFuncAttributeMaxDynamicSharedMemorySize, SMEM_BYTES);

  dcrnn_encoder_kernel<<<2 * BB, 256, SMEM_BYTES, stream>>>(
      xin, slen, p0, p1, wbase, last);

  fc_kernel<<<BB, NN, 0, stream>>>(last, (const float*)d_in[20],
                                   (const float*)d_in[21], (float*)d_out);
}
